// CARDAttention_8151847928123
// MI455X (gfx1250) — matrix operand order repaired and checked by reading, not compiled, not run on hardware
//
#include <hip/hip_runtime.h>
#include <hip/hip_bf16.h>
#include <math.h>

// ---------------------------------------------------------------------------
// CARD attention block for MI455X (gfx1250, wave32, WMMA).
// GEMM hot loop: pre-packed bf16 fragments, 12 b128 loads + 8 wmma per k-step
// (wave tile 32Mx64N). Attention stages E/q/k/v into LDS with
// global_load_async_to_lds_b128 (ASYNCcnt). Batchnorm fused into packing;
// GELU + bf16 A-layout packing fused into the FF1 epilogue.
// ---------------------------------------------------------------------------

#define B_  16
#define N_  32
#define H_  64
#define D_  256
#define DFF_ 1024
#define HEADS_ 8
#define HD_ 32
#define ROWS_ (B_ * N_ * H_)      // 32768
#define BNROWS_ (B_ * N_)         // 512
#define MCHUNK_ 4096

typedef __attribute__((ext_vector_type(16))) __bf16         v16bf;
typedef __attribute__((ext_vector_type(16))) unsigned short v16u;
typedef __attribute__((ext_vector_type(8)))  float          v8f;

__device__ __forceinline__ unsigned short f2bf(float f) {
  unsigned int u = __float_as_uint(f);
  u += 0x7fffu + ((u >> 16) & 1u);          // round-to-nearest-even
  return (unsigned short)(u >> 16);
}

// --- CDNA5 async global->LDS copy (ASYNCcnt), see CDNA5 ISA
__device__ __forceinline__ void async_copy_b128(void* lds_dst, const float* gsrc) {
  unsigned lds = (unsigned)(unsigned long long)lds_dst;   // LDS byte address
  asm volatile("global_load_async_to_lds_b128 %0, %1, off"
               :: "v"(lds), "v"(gsrc) : "memory");
}
__device__ __forceinline__ void async_wait0() {
  asm volatile("s_wait_asynccnt 0x0" ::: "memory");
}

// A/B fragment K-map (16-bit 16x32 A layout, and the mirrored B layout):
//   lane<16 : halves 0..7 -> K=0..7,   halves 8..15 -> K=16..23
//   lane>=16: halves 0..7 -> K=8..15,  halves 8..15 -> K=24..31
// B tile (32K x 16N): lane L holds col (L&15); halves e hold K=(e&7)+((L>=16)?8:0)+((e>=8)?16:0).

// ---------------------------------------------------------------------------
// Pack row-major f32 weight (K x N) into B-fragment tiles (32x16), bf16.
// ---------------------------------------------------------------------------
__global__ void pack_weight_bf16(const float* __restrict__ W,
                                 unsigned short* __restrict__ P,
                                 int K, int N) {
  int idx = blockIdx.x * blockDim.x + threadIdx.x;
  if (idx >= K * N) return;
  int tile = idx >> 9;
  int r    = idx & 511;
  int L    = r >> 4;
  int e    = r & 15;
  int tilesN = N >> 4;
  int k0 = (tile / tilesN) << 5;
  int n0 = (tile % tilesN) << 4;
  int k = k0 + (e & 7) + ((L >> 4) << 3) + ((e & 8) << 1);
  int n = n0 + (L & 15);
  P[idx] = f2bf(W[(size_t)k * N + n]);
}

// ---------------------------------------------------------------------------
// Pack row-major f32 activations (M x K) into A-fragment tiles (16M x 32K),
// bf16, optionally applying per-channel batchnorm scale/shift.
// ---------------------------------------------------------------------------
template<bool NORM>
__global__ void pack_act_bf16(const float* __restrict__ X,
                              unsigned short* __restrict__ P,
                              const float* __restrict__ nscale,
                              const float* __restrict__ nshift,
                              int M, int K) {
  int idx = blockIdx.x * blockDim.x + threadIdx.x;
  if (idx >= M * K) return;
  int tile = idx >> 9;
  int r    = idx & 511;
  int L    = r >> 4;
  int e    = r & 15;
  int tilesK = K >> 5;
  int m0 = (tile / tilesK) << 4;
  int k0 = (tile % tilesK) << 5;
  int hi = L >> 4;
  int row = m0 + (L & 15);
  int kk  = k0 + (e & 7) + (hi << 3) + ((e & 8) << 1);
  float x = X[(size_t)row * K + kk];
  if (NORM) x = x * nscale[kk] + nshift[kk];
  P[idx] = f2bf(x);
}

// ---------------------------------------------------------------------------
// WMMA GEMM on pre-packed bf16 fragments.
//   block = 128 threads = 4 waves; block tile 128(M) x 64(N);
//   wave tile 32(M) x 64(N): 2 A frags + 4 B frags feed 8 wmma per k-step.
// FLAGS bit1: exact GELU on output
// FLAGS bit2: add res[row,col]
// FLAGS bit3: accumulate into existing Cf
// FLAGS bit4: store output as packed bf16 A-layout into Cp (K-dim = N)
// ---------------------------------------------------------------------------
template<int FLAGS>
__global__ void __launch_bounds__(128)
gemm_bf16_wmma(const unsigned short* __restrict__ Ap,
               const unsigned short* __restrict__ Bp,
               const float* __restrict__ bias,
               float* __restrict__ Cf,
               unsigned short* __restrict__ Cp,
               const float* __restrict__ res,
               int M, int N, int K)
{
  const int lane = threadIdx.x & 31;
  const int wave = threadIdx.x >> 5;
  const int m0 = (blockIdx.x << 7) + (wave << 5);
  const int n0 = blockIdx.y << 6;
  const int hi = lane >> 4;
  const int colbase = n0 + (lane & 15);
  const int tilesN = N >> 4;
  const int tilesK = K >> 5;

  v8f acc[4][2];
  #pragma unroll
  for (int t = 0; t < 4; ++t) {
    const float bv = bias[colbase + t * 16];
    #pragma unroll
    for (int h = 0; h < 2; ++h)
      #pragma unroll
      for (int r = 0; r < 8; ++r) acc[t][h][r] = bv;
  }

  const unsigned short* __restrict__ arow0 =
      Ap + ((size_t)(m0 >> 4) * tilesK) * 512 + lane * 16;
  const unsigned short* __restrict__ arow1 = arow0 + (size_t)tilesK * 512;
  const unsigned short* __restrict__ brow =
      Bp + ((size_t)(n0 >> 4)) * 512 + lane * 16;

  #pragma unroll 2
  for (int k = 0; k < K; k += 32) {
    const size_t ko = (size_t)(k >> 5) * 512;
    // issue all fragment loads first so waits can be staggered against wmma
    const v16u av0 = *reinterpret_cast<const v16u*>(arow0 + ko);
    const v16u av1 = *reinterpret_cast<const v16u*>(arow1 + ko);
    const unsigned short* bk = brow + (size_t)(k >> 5) * tilesN * 512;
    v16u bw[4];
    #pragma unroll
    for (int t = 0; t < 4; ++t)
      bw[t] = *reinterpret_cast<const v16u*>(bk + (size_t)t * 512);
    #pragma unroll
    for (int t = 0; t < 4; ++t) {
      acc[t][0] = __builtin_amdgcn_wmma_f32_16x16x32_bf16(
          false, __builtin_bit_cast(v16bf, av0),
          false, __builtin_bit_cast(v16bf, bw[t]),
          (short)0, acc[t][0], false, false);
      acc[t][1] = __builtin_amdgcn_wmma_f32_16x16x32_bf16(
          false, __builtin_bit_cast(v16bf, av1),
          false, __builtin_bit_cast(v16bf, bw[t]),
          (short)0, acc[t][1], false, false);
    }
  }

  // Epilogue. C/D layout: VGPR r, lanes 0-15 -> M=base+r, lanes 16-31 -> M=base+r+8.
  #pragma unroll
  for (int t = 0; t < 4; ++t) {
    const int col = colbase + t * 16;
    #pragma unroll
    for (int h = 0; h < 2; ++h) {
      #pragma unroll
      for (int r = 0; r < 8; ++r) {
        const int rr = m0 + h * 16 + r + hi * 8;
        float v = acc[t][h][r];
        if (FLAGS & 2) v = 0.5f * v * (1.0f + erff(v * 0.70710678118f));
        if (FLAGS & 16) {
          // scatter into the consumer's A-fragment layout (consumer K-dim = N)
          const int ctile = (rr >> 4) * (N >> 5) + (col >> 5);
          const int chi   = (col >> 3) & 1;
          const int ce    = (col & 7) + ((col & 16) >> 1);
          const int clane = (rr & 15) + (chi << 4);
          Cp[(size_t)ctile * 512 + clane * 16 + ce] = f2bf(v);
        } else {
          const size_t idx = (size_t)rr * N + col;
          if (FLAGS & 4) v += res[idx];
          if (FLAGS & 8) v += Cf[idx];
          Cf[idx] = v;
        }
      }
    }
  }
}

// ---------------------------------------------------------------------------
// Per-(bn, head) dual-axis attention (VALU f32 in LDS; softmax-bound).
// Inputs staged with CDNA5 async global->LDS copies. Results written in the
// merged-heads layout: t=(head&1)*64+row; Hidx=t>>1; c=(t&1)*128+(head>>1)*32+d.
// ---------------------------------------------------------------------------
__device__ __forceinline__ void write_merged(float* __restrict__ dst, int bn,
                                             int head, int row, int d, float val) {
  const int t  = ((head & 1) << 6) + row;
  const int Hi = t >> 1;
  const int c  = ((t & 1) << 7) + ((head >> 1) << 5) + d;
  dst[((size_t)bn * H_ + Hi) * D_ + c] = val;
}

__global__ void __launch_bounds__(256)
card_attention_kernel(const float* __restrict__ qkv, const float* __restrict__ ema,
                      float* __restrict__ o1, float* __restrict__ o2)
{
  __shared__ float smem[4096 + 5 * 2048];   // 57344 B
  float* Esc = smem;            // 64x64: EMA matrix, later reused for scores
  float* qs  = smem + 4096;     // 64x32
  float* ks  = smem + 6144;
  float* vs  = smem + 8192;
  float* eq  = smem + 10240;
  float* ek  = smem + 12288;

  const int bn   = blockIdx.x;
  const int head = blockIdx.y;
  const int tid  = threadIdx.x;

  // ---- async global -> LDS staging (tracked by ASYNCcnt) ----
  #pragma unroll
  for (int j = 0; j < 4; ++j) {             // E: 1024 float4
    const int i4 = tid + j * 256;
    async_copy_b128(&Esc[i4 * 4], ema + (size_t)i4 * 4);
  }
  #pragma unroll
  for (int j = 0; j < 2; ++j) {             // q/k/v: 512 float4 each
    const int i4 = tid + j * 256;
    const int a = i4 >> 3, seg = i4 & 7;
    const size_t g = ((size_t)bn * H_ + a) * (3 * D_) + head * HD_ + seg * 4;
    const int loff = a * HD_ + seg * 4;
    async_copy_b128(&qs[loff], qkv + g);
    async_copy_b128(&ks[loff], qkv + g + D_);
    async_copy_b128(&vs[loff], qkv + g + 2 * D_);
  }
  async_wait0();
  __syncthreads();

  for (int i = tid; i < 2048; i += 256) {      // eq/ek = EMA @ q / EMA @ k
    const int g = i >> 5, d = i & 31;
    float s0 = 0.f, s1 = 0.f;
    for (int a = 0; a < H_; ++a) {
      const float e = Esc[g * H_ + a];
      s0 += e * qs[a * HD_ + d];
      s1 += e * ks[a * HD_ + d];
    }
    eq[i] = s0; ek[i] = s1;
  }
  __syncthreads();

  // token-axis attention (64x64 scores, scale sqrt(hd))
  for (int i = tid; i < 4096; i += 256) {
    const int e = i >> 6, f = i & 63;
    float s = 0.f;
    for (int d = 0; d < HD_; ++d) s += eq[e * HD_ + d] * ek[f * HD_ + d];
    Esc[i] = s * 5.65685424949f;
  }
  __syncthreads();
  if (tid < 64) {
    float* row = Esc + tid * 64;
    float m = row[0];
    for (int f = 1; f < 64; ++f) m = fmaxf(m, row[f]);
    float s = 0.f;
    for (int f = 0; f < 64; ++f) { const float e = __expf(row[f] - m); row[f] = e; s += e; }
    const float inv = 1.f / s;
    for (int f = 0; f < 64; ++f) row[f] *= inv;
  }
  __syncthreads();
  for (int i = tid; i < 2048; i += 256) {
    const int e = i >> 5, d = i & 31;
    float s = 0.f;
    for (int f = 0; f < H_; ++f) s += Esc[e * 64 + f] * vs[f * HD_ + d];
    write_merged(o1, bn, head, e, d, s);
  }
  __syncthreads();

  // hidden-axis attention (32x32 scores, scale sqrt(H))
  for (int i = tid; i < 1024; i += 256) {
    const int e = i >> 5, f = i & 31;
    float s = 0.f;
    for (int a = 0; a < H_; ++a) s += qs[a * HD_ + e] * ks[a * HD_ + f];
    Esc[e * 32 + f] = s * 8.0f;
  }
  __syncthreads();
  if (tid < 32) {
    float* row = Esc + tid * 32;
    float m = row[0];
    for (int f = 1; f < 32; ++f) m = fmaxf(m, row[f]);
    float s = 0.f;
    for (int f = 0; f < 32; ++f) { const float e = __expf(row[f] - m); row[f] = e; s += e; }
    const float inv = 1.f / s;
    for (int f = 0; f < 32; ++f) row[f] *= inv;
  }
  __syncthreads();
  for (int i = tid; i < 2048; i += 256) {
    const int a = i >> 5, e = i & 31;
    float s = 0.f;
    for (int f = 0; f < HD_; ++f) s += Esc[e * 32 + f] * vs[a * HD_ + f];
    write_merged(o2, bn, head, a, e, s);
  }
}

// ---------------------------------------------------------------------------
// Deterministic batchnorm stats (coalesced partials, no float atomics).
// ---------------------------------------------------------------------------
__global__ void __launch_bounds__(256)
bn_stats_partial(const float* __restrict__ X, float* __restrict__ partial) {
  const int ch = threadIdx.x;
  const int r0 = blockIdx.x * 256;
  float s = 0.f, s2 = 0.f;
  for (int j = 0; j < 256; ++j) {
    const float x = X[((size_t)(r0 + j)) * D_ + ch];
    s += x; s2 += x * x;
  }
  partial[blockIdx.x * 512 + ch]       = s;
  partial[blockIdx.x * 512 + 256 + ch] = s2;
}

__global__ void __launch_bounds__(256)
bn_stats_finalize(const float* __restrict__ partial,
                  const float* __restrict__ g, const float* __restrict__ be,
                  float* __restrict__ nscale, float* __restrict__ nshift) {
  const int ch = threadIdx.x;
  float s = 0.f, s2 = 0.f;
  for (int b = 0; b < 128; ++b) {
    s  += partial[b * 512 + ch];
    s2 += partial[b * 512 + 256 + ch];
  }
  const float invN = 1.0f / (float)ROWS_;
  const float m   = s * invN;
  const float var = s2 * invN - m * m;
  const float rs  = rsqrtf(var + 1e-5f);
  const float sc  = g[ch] * rs;
  nscale[ch] = sc;
  nshift[ch] = be[ch] - m * sc;
}

__global__ void __launch_bounds__(256)
bn_apply(const float* __restrict__ X, const float* __restrict__ nscale,
         const float* __restrict__ nshift, float* __restrict__ out, int n) {
  const int i = blockIdx.x * blockDim.x + threadIdx.x;
  if (i < n) {
    const int ch = i & (D_ - 1);
    out[i] = X[i] * nscale[ch] + nshift[ch];
  }
}

// ---------------------------------------------------------------------------
extern "C" void kernel_launch(void* const* d_in, const int* in_sizes, int n_in,
                              void* d_out, int out_size, void* d_ws, size_t ws_size,
                              hipStream_t stream) {
  (void)in_sizes; (void)n_in; (void)out_size; (void)ws_size;

  const float* src  = (const float*)d_in[0];
  const float* ema  = (const float*)d_in[1];
  const float* Wqkv = (const float*)d_in[2];
  const float* bqkv = (const float*)d_in[3];
  const float* W1a  = (const float*)d_in[4];
  const float* b1a  = (const float*)d_in[5];
  const float* W2a  = (const float*)d_in[6];
  const float* b2a  = (const float*)d_in[7];
  const float* W1b  = (const float*)d_in[8];
  const float* b1b  = (const float*)d_in[9];
  const float* W2b  = (const float*)d_in[10];
  const float* b2b  = (const float*)d_in[11];
  const float* g1 = (const float*)d_in[12]; const float* be1 = (const float*)d_in[13];
  const float* g2 = (const float*)d_in[14]; const float* be2 = (const float*)d_in[15];
  const float* g3 = (const float*)d_in[16]; const float* be3 = (const float*)d_in[17];
  float* out = (float*)d_out;

  // --- workspace carve-up ---
  char* ws = (char*)d_ws;
  size_t off = 0;
  auto carve = [&](size_t bytes) { char* p = ws + off; off += (bytes + 255) & ~(size_t)255; return p; };
  float* qkv  = (float*)carve((size_t)ROWS_ * 768 * 4);                 // 96 MB
  float* o1   = (float*)carve((size_t)ROWS_ * D_ * 4);                  // 32 MB
  float* o2   = (float*)carve((size_t)ROWS_ * D_ * 4);                  // 32 MB
  unsigned short* apack = (unsigned short*)carve((size_t)ROWS_ * D_ * 2);     // 16 MB
  unsigned short* hpack = (unsigned short*)carve((size_t)MCHUNK_ * DFF_ * 2); // 8 MB
  unsigned short* pWqkv = (unsigned short*)carve((size_t)D_ * 768 * 2);
  unsigned short* pW1a  = (unsigned short*)carve((size_t)D_ * DFF_ * 2);
  unsigned short* pW2a  = (unsigned short*)carve((size_t)DFF_ * D_ * 2);
  unsigned short* pW1b  = (unsigned short*)carve((size_t)D_ * DFF_ * 2);
  unsigned short* pW2b  = (unsigned short*)carve((size_t)DFF_ * D_ * 2);
  float* partial = (float*)carve(128 * 512 * 4);
  float* nsc1 = (float*)carve(D_ * 4); float* nsh1 = (float*)carve(D_ * 4);
  float* nsc2 = (float*)carve(D_ * 4); float* nsh2 = (float*)carve(D_ * 4);
  float* nsc3 = (float*)carve(D_ * 4); float* nsh3 = (float*)carve(D_ * 4);
  float* pre = out;   // residual-sum lives in d_out; final BN is in-place

  // --- 1. pack weights (bf16 B-fragment layout) ---
  auto packW = [&](const float* W, unsigned short* P, int K, int N) {
    const int tot = K * N;
    hipLaunchKernelGGL(pack_weight_bf16, dim3((tot + 255) / 256), dim3(256), 0, stream, W, P, K, N);
  };
  packW(Wqkv, pWqkv, D_, 768);
  packW(W1a, pW1a, D_, DFF_);
  packW(W2a, pW2a, DFF_, D_);
  packW(W1b, pW1b, D_, DFF_);
  packW(W2b, pW2b, DFF_, D_);

  // --- 2. QKV projection ---
  hipLaunchKernelGGL((pack_act_bf16<false>), dim3((ROWS_ * D_ + 255) / 256), dim3(256), 0, stream,
                     src, apack, nsc1, nsh1, ROWS_, D_);
  hipLaunchKernelGGL((gemm_bf16_wmma<0>), dim3(ROWS_ / 128, 768 / 64), dim3(128), 0, stream,
                     apack, pWqkv, bqkv, qkv, hpack, src, ROWS_, 768, D_);

  // --- 3. dual-axis attention -> merged-head o1 / o2 ---
  hipLaunchKernelGGL(card_attention_kernel, dim3(BNROWS_, HEADS_), dim3(256), 0, stream,
                     qkv, ema, o1, o2);

  auto bn_stats = [&](const float* X, const float* g, const float* be,
                      float* nsc, float* nsh) {
    hipLaunchKernelGGL(bn_stats_partial, dim3(128), dim3(256), 0, stream, X, partial);
    hipLaunchKernelGGL(bn_stats_finalize, dim3(1), dim3(256), 0, stream, partial, g, be, nsc, nsh);
  };

  const int tilesPerChunkA = (MCHUNK_ / 16) * (D_ / 32);   // A-layout tiles per M-chunk, K=256

  // --- 4. FFN A: pre = src + gelu(BN1(o1)@W1a+b1a)@W2a+b2a ---
  bn_stats(o1, g1, be1, nsc1, nsh1);
  hipLaunchKernelGGL((pack_act_bf16<true>), dim3((ROWS_ * D_ + 255) / 256), dim3(256), 0, stream,
                     o1, apack, nsc1, nsh1, ROWS_, D_);
  for (int chunk = 0; chunk < ROWS_ / MCHUNK_; ++chunk) {
    const size_t r0 = (size_t)chunk * MCHUNK_;
    hipLaunchKernelGGL((gemm_bf16_wmma<2 | 16>), dim3(MCHUNK_ / 128, DFF_ / 64), dim3(128), 0, stream,
                       apack + (size_t)chunk * tilesPerChunkA * 512, pW1a, b1a,
                       pre, hpack, src, MCHUNK_, DFF_, D_);
    hipLaunchKernelGGL((gemm_bf16_wmma<4>), dim3(MCHUNK_ / 128, D_ / 64), dim3(128), 0, stream,
                       hpack, pW2a, b2a, pre + r0 * D_, hpack, src + r0 * D_,
                       MCHUNK_, D_, DFF_);
  }

  // --- 5. FFN B: pre += gelu(BN2(o2)@W1b+b1b)@W2b+b2b ---
  bn_stats(o2, g2, be2, nsc2, nsh2);
  hipLaunchKernelGGL((pack_act_bf16<true>), dim3((ROWS_ * D_ + 255) / 256), dim3(256), 0, stream,
                     o2, apack, nsc2, nsh2, ROWS_, D_);
  for (int chunk = 0; chunk < ROWS_ / MCHUNK_; ++chunk) {
    const size_t r0 = (size_t)chunk * MCHUNK_;
    hipLaunchKernelGGL((gemm_bf16_wmma<2 | 16>), dim3(MCHUNK_ / 128, DFF_ / 64), dim3(128), 0, stream,
                       apack + (size_t)chunk * tilesPerChunkA * 512, pW1b, b1b,
                       pre, hpack, src, MCHUNK_, DFF_, D_);
    hipLaunchKernelGGL((gemm_bf16_wmma<8>), dim3(MCHUNK_ / 128, D_ / 64), dim3(128), 0, stream,
                       hpack, pW2b, b2b, pre + r0 * D_, hpack, src + r0 * D_,
                       MCHUNK_, D_, DFF_);
  }

  // --- 6. final batchnorm into d_out ---
  bn_stats(pre, g3, be3, nsc3, nsh3);
  const int ntot = ROWS_ * D_;
  hipLaunchKernelGGL(bn_apply, dim3((ntot + 255) / 256), dim3(256), 0, stream,
                     pre, nsc3, nsh3, out, ntot);
}
